// XCAf_15564961480777
// MI455X (gfx1250) — matrix operand order repaired and checked by reading, not compiled, not run on hardware
//
#include <hip/hip_runtime.h>

// ---------------------------------------------------------------------------
// XCA (cross-covariance attention) for MI455X / gfx1250.
// All GEMMs run as NT bf16 WMMA (v_wmma_f32_16x16x32_bf16) with fp32
// accumulation and fully-unrolled software-pipelined fragment loads.
// L2-normalize and softmax stay in fp32.
// ---------------------------------------------------------------------------

typedef __bf16 bf16_t;
typedef __attribute__((ext_vector_type(16))) __bf16 v16bf;
typedef __attribute__((ext_vector_type(8)))  __bf16 bf16x8;
typedef __attribute__((ext_vector_type(4)))  __bf16 bf16x4;
typedef __attribute__((ext_vector_type(8)))  float  v8f;

#define NUM_B 4
#define NUM_N 4096
#define NUM_C 768
#define NUM_H 8
#define CH    96
#define KB    256
#define M_ALL (NUM_B * NUM_N)            // 16384 rows (b,n)
#define ROWS_BHN (NUM_B * NUM_H * NUM_N) // 131072 rows (b,h,n)

// ---------------------------------------------------------------------------
// Pipelined NT GEMM: D[16*MT x 16*NT] += A[16*MT x K] * Bt[16*NT x K]^T
// K is a template constant so the pipelined loop fully unrolls: no
// loop-carried accumulator copies, fragments become distinct SSA values.
// A fragment (16-bit A 16x32): lane (m=lane&15, h=lane>>4) holds
//   K {k0+8h..+7} in VGPR0-3 and {k0+16+8h..} in VGPR4-7.
// B fragment (16-bit B 32x16): lane (n=lane&15, h=lane>>4) holds
//   K {k0+8h..+7} in VGPR0-3 and {k0+16+8h..} in VGPR4-7.
// ---------------------------------------------------------------------------
template <int MT, int NT, int K>
__device__ __forceinline__ void gemm_nt(const bf16_t* __restrict__ A, int lda,
                                        const bf16_t* __restrict__ Bt, int ldb,
                                        v8f (&acc)[MT][NT], int lane) {
  const int l15 = lane & 15;
  const int h = lane >> 4;
  const bf16_t* ap = A + (long)l15 * lda + 8 * h;
  const bf16_t* bp = Bt + (long)l15 * ldb + 8 * h;

  v16bf aP[MT], bP[NT], aQ[MT], bQ[NT];

  auto load_frag = [&](v16bf (&av)[MT], v16bf (&bv)[NT], int k0) {
#pragma unroll
    for (int m = 0; m < MT; ++m) {
      const bf16_t* p = ap + (long)m * 16 * lda + k0;
      ((bf16x8*)&av[m])[0] = *(const bf16x8*)(p);
      ((bf16x8*)&av[m])[1] = *(const bf16x8*)(p + 16);
    }
#pragma unroll
    for (int t = 0; t < NT; ++t) {
      const bf16_t* p = bp + (long)t * 16 * ldb + k0;
      ((bf16x8*)&bv[t])[0] = *(const bf16x8*)(p);
      ((bf16x8*)&bv[t])[1] = *(const bf16x8*)(p + 16);
    }
  };
  auto mma = [&](v16bf (&av)[MT], v16bf (&bv)[NT]) {
#pragma unroll
    for (int m = 0; m < MT; ++m)
#pragma unroll
      for (int t = 0; t < NT; ++t)
        acc[m][t] = __builtin_amdgcn_wmma_f32_16x16x32_bf16(
            false, av[m], false, bv[t], (short)0, acc[m][t], false, false);
  };

  constexpr int steps = K >> 5;
  load_frag(aP, bP, 0);
  int s = 0;
#pragma unroll
  for (; s + 2 < steps; s += 2) {
    load_frag(aQ, bQ, (s + 1) << 5);
    mma(aP, bP);
    load_frag(aP, bP, (s + 2) << 5);
    mma(aQ, bQ);
  }
  if constexpr (((steps & 1) == 0)) {
    load_frag(aQ, bQ, (steps - 1) << 5);
    mma(aP, bP);
    mma(aQ, bQ);
  } else {
    mma(aP, bP);
  }
}

// ---------------------------------------------------------------------------
// fp32 -> bf16 cast, 4 elements/thread
// ---------------------------------------------------------------------------
__global__ void k_cast4(const float4* __restrict__ in,
                        bf16x4* __restrict__ out, long n4) {
  long i = (long)blockIdx.x * blockDim.x + threadIdx.x;
  if (i < n4) {
    float4 v = in[i];
    bf16x4 o = {(bf16_t)v.x, (bf16_t)v.y, (bf16_t)v.z, (bf16_t)v.w};
    out[i] = o;
  }
}

// ---------------------------------------------------------------------------
// Stage 1: qkv = x @ qkv_w^T, scattered into q/k/v as (B,H,N,Ch) bf16.
// Wave tile 32x64 (MT=2,NT=4). waves = 512*36 = 18432 -> 2304 blocks.
// ---------------------------------------------------------------------------
__global__ void __launch_bounds__(256) k_qkv(const bf16_t* __restrict__ xb,
                                             const bf16_t* __restrict__ wqb,
                                             bf16_t* __restrict__ qn,
                                             bf16_t* __restrict__ kn,
                                             bf16_t* __restrict__ vn) {
  int wid = (blockIdx.x * 256 + threadIdx.x) >> 5;
  int lane = threadIdx.x & 31;
  int nt = wid % 36, mt = wid / 36;
  const bf16_t* A = xb + (long)mt * 32 * NUM_C;
  const bf16_t* Bt = wqb + (long)nt * 64 * NUM_C;
  v8f acc[2][4] = {};
  gemm_nt<2, 4, NUM_C>(A, NUM_C, Bt, NUM_C, acc, lane);
  int mbase = 8 * (lane >> 4), ncol = lane & 15;
#pragma unroll
  for (int t = 0; t < 4; ++t) {
    int j = nt * 64 + t * 16 + ncol;          // 0..2303
    int s = j / NUM_C;                        // 0=q 1=k 2=v
    int rem = j - s * NUM_C;
    int hh = rem / CH, c = rem - hh * CH;
    bf16_t* dst = (s == 0) ? qn : ((s == 1) ? kn : vn);
#pragma unroll
    for (int mm = 0; mm < 2; ++mm)
#pragma unroll
      for (int r = 0; r < 8; ++r) {
        int mg = mt * 32 + mm * 16 + mbase + r;  // b*4096 + n
        int b = mg >> 12, n = mg & 4095;
        dst[((long)(b * NUM_H + hh) * NUM_N + n) * CH + c] =
            (bf16_t)acc[mm][t][r];
      }
  }
}

// ---------------------------------------------------------------------------
// Stage 2a: o1[(bh,n), kk] = sum_c t[(bh,n), c] * w0[kk, c]  (bf16 out)
// Wave tile 32x64. waves = 4096*4 = 16384 -> 2048 blocks. K=96.
// ---------------------------------------------------------------------------
__global__ void __launch_bounds__(256) k_bneck1(const bf16_t* __restrict__ tn,
                                                const bf16_t* __restrict__ w0b,
                                                bf16_t* __restrict__ o1) {
  int wid = (blockIdx.x * 256 + threadIdx.x) >> 5;
  int lane = threadIdx.x & 31;
  int nt = wid & 3, mt = wid >> 2;
  const bf16_t* A = tn + (long)mt * 32 * CH;
  const bf16_t* Bt = w0b + (long)nt * 64 * CH;
  v8f acc[2][4] = {};
  gemm_nt<2, 4, CH>(A, CH, Bt, CH, acc, lane);
  int mbase = 8 * (lane >> 4), ncol = lane & 15;
#pragma unroll
  for (int mm = 0; mm < 2; ++mm)
#pragma unroll
    for (int t = 0; t < 4; ++t)
#pragma unroll
      for (int r = 0; r < 8; ++r)
        o1[(long)(mt * 32 + mm * 16 + mbase + r) * KB + nt * 64 + t * 16 +
           ncol] = (bf16_t)acc[mm][t][r];
}

// ---------------------------------------------------------------------------
// Norm over n: norms[bh,kk] += sum_{n in chunk} o1[(bh,n),kk]^2
// grid = 32 bh * 16 chunks, 256 threads = kk (coalesced rows)
// ---------------------------------------------------------------------------
__global__ void k_normsq(const bf16_t* __restrict__ o1,
                         float* __restrict__ norms) {
  int bh = blockIdx.x >> 4;
  int chunk = blockIdx.x & 15;
  int kk = threadIdx.x;
  const bf16_t* base = o1 + ((long)bh * NUM_N + chunk * 256) * KB + kk;
  float s = 0.f;
  for (int n = 0; n < 256; ++n) {
    float v = (float)base[(long)n * KB];
    s += v * v;
  }
  atomicAdd(&norms[bh * KB + kk], s);
}

__global__ void k_invnorm(float* __restrict__ norms) {
  int i = blockIdx.x * 256 + threadIdx.x;  // 8192
  float s = sqrtf(norms[i]);
  norms[i] = 1.0f / fmaxf(s, 1e-12f);
}

// 8 contiguous elements per thread; kk runs over the contiguous axis.
__global__ void k_scale(bf16_t* __restrict__ o1, const float* __restrict__ inv) {
  long i = ((long)blockIdx.x * 256 + threadIdx.x) * 8;  // over 131072*256
  int kk = (int)(i & (KB - 1));
  int bh = (int)((i >> 8) >> 12);  // row / 4096
  const float* ip = inv + bh * KB + kk;
  bf16x8 v = *(bf16x8*)(o1 + i);
  bf16x8 o;
#pragma unroll
  for (int e = 0; e < 8; ++e) o[e] = (bf16_t)((float)v[e] * ip[e]);
  *(bf16x8*)(o1 + i) = o;
}

// ---------------------------------------------------------------------------
// Stage 2b: out2[(bh,c), n] = sum_kk w1[c,kk] * o1n[(bh,n),kk] -> (B,H,Ch,N)
// per bh: M=96 (3 mt32), N=4096 (64 nt64). waves = 32*192 = 6144 -> 768.
// ---------------------------------------------------------------------------
__global__ void __launch_bounds__(256) k_bneck2(const bf16_t* __restrict__ w1b,
                                                const bf16_t* __restrict__ o1,
                                                bf16_t* __restrict__ out2) {
  int wid = (blockIdx.x * 256 + threadIdx.x) >> 5;
  int lane = threadIdx.x & 31;
  int bh = wid / 192, rr = wid % 192;
  int mt = rr >> 6, nt = rr & 63;
  const bf16_t* A = w1b + (long)mt * 32 * KB;
  const bf16_t* Bt = o1 + (long)bh * NUM_N * KB + (long)nt * 64 * KB;
  v8f acc[2][4] = {};
  gemm_nt<2, 4, KB>(A, KB, Bt, KB, acc, lane);
  int mbase = 8 * (lane >> 4), ncol = lane & 15;
#pragma unroll
  for (int mm = 0; mm < 2; ++mm)
#pragma unroll
    for (int t = 0; t < 4; ++t) {
      int n = nt * 64 + t * 16 + ncol;
#pragma unroll
      for (int r = 0; r < 8; ++r) {
        int c = mt * 32 + mm * 16 + mbase + r;
        out2[((long)bh * CH + c) * NUM_N + n] = (bf16_t)acc[mm][t][r];
      }
    }
}

// ---------------------------------------------------------------------------
// Stage 3: attn[bh,c,d] += sum_{n in Ksplit} q2[bh,c,n]*k2[bh,d,n]
// split-K x4, fp32 atomicAdd partials. waves = 32*36*4 = 4608 -> 576 blocks.
// ---------------------------------------------------------------------------
__global__ void __launch_bounds__(256) k_attn(const bf16_t* __restrict__ q2,
                                              const bf16_t* __restrict__ k2,
                                              float* __restrict__ attnf) {
  int wid = (blockIdx.x * 256 + threadIdx.x) >> 5;
  int lane = threadIdx.x & 31;
  int bh = wid / 144, rr = wid % 144;
  int ks = rr / 36, r2 = rr % 36;
  int mt = r2 / 6, dt = r2 % 6;
  const bf16_t* A =
      q2 + (long)bh * CH * NUM_N + (long)mt * 16 * NUM_N + ks * 1024;
  const bf16_t* Bt =
      k2 + (long)bh * CH * NUM_N + (long)dt * 16 * NUM_N + ks * 1024;
  v8f acc[1][1] = {};
  gemm_nt<1, 1, 1024>(A, NUM_N, Bt, NUM_N, acc, lane);
  int mbase = 8 * (lane >> 4), ncol = lane & 15;
#pragma unroll
  for (int r = 0; r < 8; ++r) {
    int c = mt * 16 + mbase + r;
    int d = dt * 16 + ncol;
    atomicAdd(&attnf[(long)bh * (CH * CH) + c * CH + d], acc[0][0][r]);
  }
}

// ---------------------------------------------------------------------------
// Softmax over d (96), temperature folded in. One wave32 per row.
// rows = 3072 -> 384 blocks of 8 waves.
// ---------------------------------------------------------------------------
__global__ void k_softmax(const float* __restrict__ attnf,
                          const float* __restrict__ temp,
                          bf16_t* __restrict__ attnb) {
  int row = blockIdx.x * 8 + (threadIdx.x >> 5);
  int lane = threadIdx.x & 31;
  float ts = temp[(row / CH) & 7];
  const float* p = attnf + (long)row * CH;
  float v0 = p[lane] * ts, v1 = p[lane + 32] * ts, v2 = p[lane + 64] * ts;
  float m = fmaxf(v0, fmaxf(v1, v2));
  for (int o = 16; o; o >>= 1) m = fmaxf(m, __shfl_xor(m, o, 32));
  float e0 = __expf(v0 - m), e1 = __expf(v1 - m), e2 = __expf(v2 - m);
  float s = e0 + e1 + e2;
  for (int o = 16; o; o >>= 1) s += __shfl_xor(s, o, 32);
  float inv = 1.0f / s;
  bf16_t* q = attnb + (long)row * CH;
  q[lane] = (bf16_t)(e0 * inv);
  q[lane + 32] = (bf16_t)(e1 * inv);
  q[lane + 64] = (bf16_t)(e2 * inv);
}

// ---------------------------------------------------------------------------
// Stage 4: out^T[(bh,n), c] = sum_d v[(bh,n),d] * attn[bh,c,d]
// per bh: M=4096 (128 mt32), N=96 (2 nt48, NT=3). waves = 32*256 = 8192.
// ---------------------------------------------------------------------------
__global__ void __launch_bounds__(256) k_av(const bf16_t* __restrict__ vn,
                                            const bf16_t* __restrict__ attnb,
                                            bf16_t* __restrict__ obuf) {
  int wid = (blockIdx.x * 256 + threadIdx.x) >> 5;
  int lane = threadIdx.x & 31;
  int bh = wid / 256, rr = wid % 256;
  int mt = rr >> 1, ct = rr & 1;
  const bf16_t* A = vn + (long)bh * NUM_N * CH + (long)mt * 32 * CH;
  const bf16_t* Bt = attnb + (long)bh * (CH * CH) + (long)ct * 48 * CH;
  v8f acc[2][3] = {};
  gemm_nt<2, 3, CH>(A, CH, Bt, CH, acc, lane);
  int b = bh >> 3, hh = bh & 7;
  int mbase = 8 * (lane >> 4), ncol = lane & 15;
#pragma unroll
  for (int mm = 0; mm < 2; ++mm)
#pragma unroll
    for (int t = 0; t < 3; ++t) {
      int c = ct * 48 + t * 16 + ncol;
#pragma unroll
      for (int r = 0; r < 8; ++r) {
        int n = mt * 32 + mm * 16 + mbase + r;
        obuf[((long)(b * NUM_N + n)) * NUM_C + hh * CH + c] =
            (bf16_t)acc[mm][t][r];
      }
    }
}

// ---------------------------------------------------------------------------
// Stage 5: out = obuf @ proj_w^T + proj_b  (fp32 output)
// Wave tile 32x64. waves = 512*12 = 6144 -> 768 blocks.
// ---------------------------------------------------------------------------
__global__ void __launch_bounds__(256) k_proj(const bf16_t* __restrict__ obuf,
                                              const bf16_t* __restrict__ wpb,
                                              const float* __restrict__ bias,
                                              float* __restrict__ out) {
  int wid = (blockIdx.x * 256 + threadIdx.x) >> 5;
  int lane = threadIdx.x & 31;
  int nt = wid % 12, mt = wid / 12;
  const bf16_t* A = obuf + (long)mt * 32 * NUM_C;
  const bf16_t* Bt = wpb + (long)nt * 64 * NUM_C;
  v8f acc[2][4] = {};
  gemm_nt<2, 4, NUM_C>(A, NUM_C, Bt, NUM_C, acc, lane);
  int mbase = 8 * (lane >> 4), ncol = lane & 15;
#pragma unroll
  for (int t = 0; t < 4; ++t) {
    int j = nt * 64 + t * 16 + ncol;
    float bj = bias[j];
#pragma unroll
    for (int mm = 0; mm < 2; ++mm)
#pragma unroll
      for (int r = 0; r < 8; ++r) {
        int m = mt * 32 + mm * 16 + mbase + r;
        out[(long)m * NUM_C + j] = acc[mm][t][r] + bj;
      }
  }
}

// ---------------------------------------------------------------------------
extern "C" void kernel_launch(void* const* d_in, const int* in_sizes, int n_in,
                              void* d_out, int out_size, void* d_ws,
                              size_t ws_size, hipStream_t stream) {
  (void)in_sizes; (void)n_in; (void)out_size; (void)ws_size;
  const float* x      = (const float*)d_in[0];
  const float* qkv_w  = (const float*)d_in[1];
  const float* proj_w = (const float*)d_in[2];
  const float* proj_b = (const float*)d_in[3];
  const float* temp   = (const float*)d_in[4];
  const float* w0     = (const float*)d_in[5];
  const float* w1     = (const float*)d_in[6];
  float* out = (float*)d_out;

  char* base = (char*)d_ws;
  size_t off = 0;
  auto alloc = [&](size_t bytes) -> char* {
    char* p = base + off;
    off += (bytes + 255) & ~(size_t)255;
    return p;
  };

  bf16_t* xb   = (bf16_t*)alloc((size_t)M_ALL * NUM_C * 2);      // 24 MB
  bf16_t* wqb  = (bf16_t*)alloc((size_t)3 * NUM_C * NUM_C * 2);  // 3.4 MB
  bf16_t* wpb  = (bf16_t*)alloc((size_t)NUM_C * NUM_C * 2);      // 1.1 MB
  bf16_t* w0b  = (bf16_t*)alloc((size_t)KB * CH * 2);
  bf16_t* w1b  = (bf16_t*)alloc((size_t)CH * KB * 2);
  bf16_t* qn   = (bf16_t*)alloc((size_t)ROWS_BHN * CH * 2);      // 24 MB
  bf16_t* kn   = (bf16_t*)alloc((size_t)ROWS_BHN * CH * 2);
  bf16_t* vn   = (bf16_t*)alloc((size_t)ROWS_BHN * CH * 2);
  bf16_t* o1   = (bf16_t*)alloc((size_t)ROWS_BHN * KB * 2);      // 64 MB
  float*  nrm  = (float*)alloc((size_t)NUM_B * NUM_H * KB * 4);
  bf16_t* q2   = (bf16_t*)alloc((size_t)NUM_B * NUM_H * CH * NUM_N * 2);
  bf16_t* k2   = (bf16_t*)alloc((size_t)NUM_B * NUM_H * CH * NUM_N * 2);
  float*  attf = (float*)alloc((size_t)NUM_B * NUM_H * CH * CH * 4);
  bf16_t* attb = (bf16_t*)alloc((size_t)NUM_B * NUM_H * CH * CH * 2);
  bf16_t* obuf = (bf16_t*)alloc((size_t)M_ALL * NUM_C * 2);      // 24 MB

  // -- cast inputs to bf16 (4 elements/thread) ------------------------------
  k_cast4<<<(M_ALL * NUM_C / 4 + 255) / 256, 256, 0, stream>>>(
      (const float4*)x, (bf16x4*)xb, (long)M_ALL * NUM_C / 4);
  k_cast4<<<(3 * NUM_C * NUM_C / 4 + 255) / 256, 256, 0, stream>>>(
      (const float4*)qkv_w, (bf16x4*)wqb, (long)3 * NUM_C * NUM_C / 4);
  k_cast4<<<(NUM_C * NUM_C / 4 + 255) / 256, 256, 0, stream>>>(
      (const float4*)proj_w, (bf16x4*)wpb, (long)NUM_C * NUM_C / 4);
  k_cast4<<<(KB * CH / 4 + 255) / 256, 256, 0, stream>>>(
      (const float4*)w0, (bf16x4*)w0b, (long)KB * CH / 4);
  k_cast4<<<(CH * KB / 4 + 255) / 256, 256, 0, stream>>>(
      (const float4*)w1, (bf16x4*)w1b, (long)CH * KB / 4);

  // -- qkv projection -------------------------------------------------------
  k_qkv<<<2304, 256, 0, stream>>>(xb, wqb, qn, kn, vn);

  // -- bottleneck for q then k (o1 reused) ----------------------------------
  const bf16_t* bin[2] = {qn, kn};
  bf16_t* bout[2] = {q2, k2};
  for (int t = 0; t < 2; ++t) {
    k_bneck1<<<2048, 256, 0, stream>>>(bin[t], w0b, o1);
    hipMemsetAsync(nrm, 0, (size_t)NUM_B * NUM_H * KB * 4, stream);
    k_normsq<<<NUM_B * NUM_H * 16, 256, 0, stream>>>(o1, nrm);
    k_invnorm<<<NUM_B * NUM_H, 256, 0, stream>>>(nrm);
    k_scale<<<(ROWS_BHN * KB / 8) / 256, 256, 0, stream>>>(o1, nrm);
    k_bneck2<<<768, 256, 0, stream>>>(w1b, o1, bout[t]);
  }

  // -- attention ------------------------------------------------------------
  hipMemsetAsync(attf, 0, (size_t)NUM_B * NUM_H * CH * CH * 4, stream);
  k_attn<<<576, 256, 0, stream>>>(q2, k2, attf);
  k_softmax<<<384, 256, 0, stream>>>(attf, temp, attb);
  k_av<<<1024, 256, 0, stream>>>(vn, attb, obuf);

  // -- output projection ----------------------------------------------------
  k_proj<<<768, 256, 0, stream>>>(obuf, wpb, proj_b, out);
}
